// Attention_52226802319916
// MI455X (gfx1250) — matrix operand order repaired and checked by reading, not compiled, not run on hardware
//
#include <hip/hip_runtime.h>
#include <hip/hip_bf16.h>

typedef __bf16 bf16_t;
typedef __attribute__((ext_vector_type(16))) __bf16 v16bf;
typedef __attribute__((ext_vector_type(8)))  __bf16 v8bf;
typedef __attribute__((ext_vector_type(8)))  float  v8f;

// ---------------- WMMA helpers (CDNA5 wave32, 16x16x32 bf16) ----------------

__device__ __forceinline__ v8f vzero8() {
    v8f z;
#pragma unroll
    for (int i = 0; i < 8; ++i) z[i] = 0.f;
    return z;
}

__device__ __forceinline__ v8f wmma_bf16(v16bf a, v16bf b, v8f c) {
    return __builtin_amdgcn_wmma_f32_16x16x32_bf16(false, a, false, b, (short)0, c,
                                                   false, false);
}

// A fragment: 16x32 tile of row-major A (lda elements per row), starting at kbase.
// ISA layout: lane m=lane&15; lo lanes K{0..7,16..23}, hi lanes K{8..15,24..31}.
__device__ __forceinline__ v16bf load_a_frag(const bf16_t* A, int lda, int kbase) {
    int lane = threadIdx.x & 31;
    int m    = lane & 15;
    int hi   = lane >> 4;
    const bf16_t* r = A + (long)m * lda + kbase + hi * 8;
    union { v16bf v; v8bf h[2]; } u;
    u.h[0] = *(const v8bf*)(r);
    u.h[1] = *(const v8bf*)(r + 16);
    return u.v;
}

// B fragment from Bt stored N-major: Bt[n][k] (row n contiguous over k).
// ISA layout: lane n=lane&15; lo lanes K{0..7,16..23}, hi lanes K{8..15,24..31}.
__device__ __forceinline__ v16bf load_bt_frag(const bf16_t* Bt, int ldb, int kbase) {
    int lane = threadIdx.x & 31;
    int n    = lane & 15;
    int kh   = lane >> 4;
    const bf16_t* r = Bt + (long)n * ldb + kbase + kh * 8;
    union { v16bf v; v8bf h[2]; } u;
    u.h[0] = *(const v8bf*)(r);
    u.h[1] = *(const v8bf*)(r + 16);
    return u.v;
}

// 32x64 wave tile over K: acc[0..3] = rows m0..m0+15, acc[4..7] = rows m0+16..31.
// B fragment loaded once, used by both row halves (1.5 loads per WMMA).
template <int KTOT>
__device__ __forceinline__ void gemm_tile_32x64(const bf16_t* Aw, int lda,
                                                const bf16_t* Btn0, int ldb,
                                                v8f acc[8]) {
    for (int kb = 0; kb < KTOT; kb += 32) {
        v16bf a0 = load_a_frag(Aw, lda, kb);
        v16bf a1 = load_a_frag(Aw + (long)16 * lda, lda, kb);
#pragma unroll
        for (int t = 0; t < 4; ++t) {
            v16bf b = load_bt_frag(Btn0 + (long)(t * 16) * ldb, ldb, kb);
            acc[t]     = wmma_bf16(a0, b, acc[t]);
            acc[4 + t] = wmma_bf16(a1, b, acc[4 + t]);
        }
    }
}

// ---------------- conversion kernels ----------------

__global__ void cvt_kernel(const float* __restrict__ in, bf16_t* __restrict__ out, int n) {
    int i = blockIdx.x * 256 + threadIdx.x;
    if (i < n) out[i] = (bf16_t)in[i];
}

// Wsr HWIO [4,4,512,512] flat = in[r*512 + co], r = (ky*4+kx)*512+ci (8192 values).
// Produce wsrT[co][r] bf16, co-major, contiguous over r.
__global__ void wsrT_kernel(const float* __restrict__ in, bf16_t* __restrict__ out) {
    long i = (long)blockIdx.x * 256 + threadIdx.x;   // 512*8192 total
    int n = (int)(i >> 13);
    int r = (int)(i & 8191);
    out[i] = (bf16_t)in[(long)r * 512 + n];
}

// ---------------- Q projection: [16384,512] x Wq^T -> Q bf16 [32][4096][64] ---

__global__ __launch_bounds__(256) void qproj_kernel(const bf16_t* __restrict__ A,
                                                    const bf16_t* __restrict__ Bt,
                                                    const float*  __restrict__ bias,
                                                    bf16_t* __restrict__ Q) {
    int wave = threadIdx.x >> 5, lane = threadIdx.x & 31;
    int m0 = blockIdx.x * 256 + wave * 32;
    int n0 = blockIdx.y * 64;
    v8f acc[8];
#pragma unroll
    for (int t = 0; t < 8; ++t) acc[t] = vzero8();
    gemm_tile_32x64<512>(A + (long)m0 * 512, 512, Bt + (long)n0 * 512, 512, acc);

    int ncol = lane & 15, mb = (lane >> 4) * 8;
#pragma unroll
    for (int half = 0; half < 2; ++half) {
#pragma unroll
        for (int t = 0; t < 4; ++t) {
            int n = n0 + t * 16 + ncol;
            float bv = bias[n];
            int h = n >> 6, dhi = n & 63;
#pragma unroll
            for (int i = 0; i < 8; ++i) {
                int m = m0 + half * 16 + mb + i;
                int b = m >> 12, nq = m & 4095;
                Q[(((long)(b * 8 + h) * 4096 + nq) << 6) + dhi] =
                    (bf16_t)(acc[half * 4 + t][i] + bv);
            }
        }
    }
}

// ---------------- SR conv as GEMM: [1024,8192] x WsrT -> f32 [1024,512] -------

__global__ __launch_bounds__(256) void srconv_kernel(const bf16_t* __restrict__ xb,
                                                     const bf16_t* __restrict__ WsrT,
                                                     const float*  __restrict__ bsr,
                                                     float* __restrict__ Co) {
    int wave = threadIdx.x >> 5, lane = threadIdx.x & 31;
    int m0 = blockIdx.x * 256 + wave * 32;
    int n0 = blockIdx.y * 64;
    int hi = lane >> 4;
    long rb[2];
#pragma unroll
    for (int half = 0; half < 2; ++half) {
        int m  = m0 + half * 16 + (lane & 15);
        int bb = m >> 8, p = m & 255, oy = p >> 4, ox = p & 15;
        rb[half] = ((long)((bb * 64 + oy * 4) * 64 + ox * 4)) << 9;  // *512 ch
    }
    v8f acc[8];
#pragma unroll
    for (int t = 0; t < 8; ++t) acc[t] = vzero8();
    for (int kb = 0; kb < 8192; kb += 32) {
        int ky   = kb >> 11;         // 2048 = 4 px * 512 ch per input row segment
        int koff = kb & 2047;
        v16bf af[2];
#pragma unroll
        for (int half = 0; half < 2; ++half) {
            const bf16_t* r = xb + rb[half] + (long)ky * 32768 + koff + hi * 8;
            union { v16bf v; v8bf h[2]; } u;
            u.h[0] = *(const v8bf*)(r);
            u.h[1] = *(const v8bf*)(r + 16);
            af[half] = u.v;
        }
#pragma unroll
        for (int t = 0; t < 4; ++t) {
            v16bf b = load_bt_frag(WsrT + (long)(n0 + t * 16) * 8192, 8192, kb);
            acc[t]     = wmma_bf16(af[0], b, acc[t]);
            acc[4 + t] = wmma_bf16(af[1], b, acc[4 + t]);
        }
    }
    int ncol = lane & 15, mb2 = (lane >> 4) * 8;
#pragma unroll
    for (int half = 0; half < 2; ++half) {
#pragma unroll
        for (int t = 0; t < 4; ++t) {
            int n = n0 + t * 16 + ncol;
            float bv = bsr[n];
#pragma unroll
            for (int i = 0; i < 8; ++i)
                Co[(long)(m0 + half * 16 + mb2 + i) * 512 + n] =
                    acc[half * 4 + t][i] + bv;
        }
    }
}

// ---------------- LayerNorm over 512 channels, 1024 rows ----------------------

__global__ __launch_bounds__(256) void ln_kernel(const float* __restrict__ X,
                                                 const float* __restrict__ gamma,
                                                 const float* __restrict__ beta,
                                                 bf16_t* __restrict__ Y) {
    __shared__ float s1[256], s2[256];
    int row = blockIdx.x, t = threadIdx.x;
    const float* xr = X + (long)row * 512;
    float a = xr[t], b = xr[t + 256];
    s1[t] = a + b;
    s2[t] = a * a + b * b;
    __syncthreads();
    for (int off = 128; off > 0; off >>= 1) {
        if (t < off) { s1[t] += s1[t + off]; s2[t] += s2[t + off]; }
        __syncthreads();
    }
    float mu   = s1[0] * (1.f / 512.f);
    float var  = s2[0] * (1.f / 512.f) - mu * mu;
    float rstd = rsqrtf(var + 1e-5f);
    bf16_t* yr = Y + (long)row * 512;
    yr[t]       = (bf16_t)((a - mu) * rstd * gamma[t] + beta[t]);
    yr[t + 256] = (bf16_t)((b - mu) * rstd * gamma[t + 256] + beta[t + 256]);
}

// ---------------- K/V projections: [1024,512] x W^T ---------------------------
// mode 0: K -> [bh][nk][dh] ; mode 1: V -> transposed [bh][dh][nk]

__global__ __launch_bounds__(256) void kvproj_kernel(const bf16_t* __restrict__ A,
                                                     const bf16_t* __restrict__ Bt,
                                                     const float*  __restrict__ bias,
                                                     bf16_t* __restrict__ Ko,
                                                     bf16_t* __restrict__ Vt,
                                                     int mode) {
    int wave = threadIdx.x >> 5, lane = threadIdx.x & 31;
    int m0 = blockIdx.x * 256 + wave * 32;
    int n0 = blockIdx.y * 64;
    v8f acc[8];
#pragma unroll
    for (int t = 0; t < 8; ++t) acc[t] = vzero8();
    gemm_tile_32x64<512>(A + (long)m0 * 512, 512, Bt + (long)n0 * 512, 512, acc);

    int ncol = lane & 15, mb = (lane >> 4) * 8;
#pragma unroll
    for (int half = 0; half < 2; ++half) {
#pragma unroll
        for (int t = 0; t < 4; ++t) {
            int n = n0 + t * 16 + ncol;
            float bv = bias[n];
            int h = n >> 6, dhi = n & 63;
#pragma unroll
            for (int i = 0; i < 8; ++i) {
                int m = m0 + half * 16 + mb + i;
                int b = m >> 8, nk = m & 255;
                float v = acc[half * 4 + t][i] + bv;
                if (mode == 0)
                    Ko[(((long)(b * 8 + h) * 256 + nk) << 6) + dhi] = (bf16_t)v;
                else
                    Vt[(((long)(b * 8 + h) * 64 + dhi) << 8) + nk] = (bf16_t)v;
            }
        }
    }
}

// ---------------- attention: per (b,h), wave = 16 Q rows, Nk = 256 ------------

__global__ __launch_bounds__(256) void attn_kernel(const bf16_t* __restrict__ Q,
                                                   const bf16_t* __restrict__ Kb,
                                                   const bf16_t* __restrict__ Vt,
                                                   bf16_t* __restrict__ O) {
    __shared__ __align__(16) bf16_t sP[8 * 16 * 256];  // 64 KB: P fragments per wave
    int bh = blockIdx.x;                                // 0..31 = (b,h)
    int qg = blockIdx.y;                                // 0..31
    int wave = threadIdx.x >> 5, lane = threadIdx.x & 31;
    int qrow0 = qg * 128 + wave * 16;

    const bf16_t* Aq  = Q  + ((long)bh * 4096 + qrow0) * 64;
    const bf16_t* Kbh = Kb + (long)bh * 256 * 64;   // [nk][dh]
    const bf16_t* Vbh = Vt + (long)bh * 64 * 256;   // [dh][nk]

    // S = Q K^T : 16 x 256 in 16 C-fragments
    v8f s[16];
#pragma unroll
    for (int f = 0; f < 16; ++f) s[f] = vzero8();
#pragma unroll
    for (int kb = 0; kb < 64; kb += 32) {
        v16bf a = load_a_frag(Aq, 64, kb);
#pragma unroll
        for (int f = 0; f < 16; ++f) {
            v16bf b = load_bt_frag(Kbh + (long)(f * 16) * 64, 64, kb);
            s[f] = wmma_bf16(a, b, s[f]);
        }
    }

    // softmax over 256 cols; row i lives on half-wave (lo: rows 0-7, hi: 8-15)
    const float scale = 0.125f;  // dh^-0.5, dh=64
    float rinv[8];
#pragma unroll
    for (int i = 0; i < 8; ++i) {
        float mx = -3.402823466e38f;
#pragma unroll
        for (int f = 0; f < 16; ++f) {
            float v = s[f][i] * scale;
            s[f][i] = v;
            mx = fmaxf(mx, v);
        }
#pragma unroll
        for (int o = 1; o < 16; o <<= 1) mx = fmaxf(mx, __shfl_xor(mx, o, 16));
        float sm = 0.f;
#pragma unroll
        for (int f = 0; f < 16; ++f) {
            float p = __expf(s[f][i] - mx);
            s[f][i] = p;
            sm += p;
        }
#pragma unroll
        for (int o = 1; o < 16; o <<= 1) sm += __shfl_xor(sm, o, 16);
        rinv[i] = 1.f / sm;
    }

    // C-fragment layout -> A-fragment layout via LDS bounce (P as bf16)
    bf16_t* P = sP + wave * (16 * 256);
    int ncol = lane & 15, mb = (lane >> 4) * 8;
#pragma unroll
    for (int f = 0; f < 16; ++f) {
        int n = f * 16 + ncol;
#pragma unroll
        for (int i = 0; i < 8; ++i) P[(mb + i) * 256 + n] = (bf16_t)s[f][i];
    }
    __syncthreads();

    // O = P V : 16 x 64
    v8f o[4];
#pragma unroll
    for (int t = 0; t < 4; ++t) o[t] = vzero8();
    for (int kb = 0; kb < 256; kb += 32) {
        v16bf a = load_a_frag(P, 256, kb);
#pragma unroll
        for (int t = 0; t < 4; ++t) {
            v16bf b = load_bt_frag(Vbh + (long)(t * 16) * 256, 256, kb);
            o[t] = wmma_bf16(a, b, o[t]);
        }
    }

    int b = bh >> 3, h = bh & 7;
#pragma unroll
    for (int t = 0; t < 4; ++t) {
        int dhi = t * 16 + ncol;
#pragma unroll
        for (int i = 0; i < 8; ++i) {
            int m = qrow0 + mb + i;
            float val = o[t][i] * rinv[i];
            O[((long)(b * 4096 + m)) * 512 + h * 64 + dhi] = (bf16_t)val;
        }
    }
}

// ---------------- output projection: [16384,512] x Wp^T + bp -> f32 -----------

__global__ __launch_bounds__(256) void oproj_kernel(const bf16_t* __restrict__ A,
                                                    const bf16_t* __restrict__ Bt,
                                                    const float*  __restrict__ bias,
                                                    float* __restrict__ Out) {
    int wave = threadIdx.x >> 5, lane = threadIdx.x & 31;
    int m0 = blockIdx.x * 256 + wave * 32;
    int n0 = blockIdx.y * 64;
    v8f acc[8];
#pragma unroll
    for (int t = 0; t < 8; ++t) acc[t] = vzero8();
    gemm_tile_32x64<512>(A + (long)m0 * 512, 512, Bt + (long)n0 * 512, 512, acc);

    int ncol = lane & 15, mb = (lane >> 4) * 8;
#pragma unroll
    for (int half = 0; half < 2; ++half) {
#pragma unroll
        for (int t = 0; t < 4; ++t) {
            int n = n0 + t * 16 + ncol;
            float bv = bias[n];
#pragma unroll
            for (int i = 0; i < 8; ++i)
                Out[(long)(m0 + half * 16 + mb + i) * 512 + n] =
                    acc[half * 4 + t][i] + bv;
        }
    }
}

// ---------------- host launch -------------------------------------------------

extern "C" void kernel_launch(void* const* d_in, const int* in_sizes, int n_in,
                              void* d_out, int out_size, void* d_ws, size_t ws_size,
                              hipStream_t stream) {
    (void)in_sizes; (void)n_in; (void)out_size; (void)ws_size;
    const float* x    = (const float*)d_in[0];
    const float* Wq   = (const float*)d_in[1];
    const float* bq   = (const float*)d_in[2];
    const float* Wk   = (const float*)d_in[3];
    const float* bk   = (const float*)d_in[4];
    const float* Wv   = (const float*)d_in[5];
    const float* bv   = (const float*)d_in[6];
    const float* Wp   = (const float*)d_in[7];
    const float* bp   = (const float*)d_in[8];
    const float* Wsr  = (const float*)d_in[9];
    const float* bsr  = (const float*)d_in[10];
    const float* gamma= (const float*)d_in[11];
    const float* beta = (const float*)d_in[12];
    float* out = (float*)d_out;

    char* ws = (char*)d_ws;
    size_t off = 0;
    auto take = [&](size_t bytes) -> char* {
        char* p = ws + off;
        off = (off + bytes + 255) & ~(size_t)255;
        return p;
    };
    bf16_t* xb   = (bf16_t*)take(16384ull * 512 * 2);
    bf16_t* wqb  = (bf16_t*)take(512ull * 512 * 2);
    bf16_t* wkb  = (bf16_t*)take(512ull * 512 * 2);
    bf16_t* wvb  = (bf16_t*)take(512ull * 512 * 2);
    bf16_t* wpb  = (bf16_t*)take(512ull * 512 * 2);
    bf16_t* wsrT = (bf16_t*)take(512ull * 8192 * 2);
    bf16_t* qbuf = (bf16_t*)take(32ull * 4096 * 64 * 2);
    float*  cbuf = (float*) take(1024ull * 512 * 4);
    bf16_t* xkv  = (bf16_t*)take(1024ull * 512 * 2);
    bf16_t* kbuf = (bf16_t*)take(32ull * 256 * 64 * 2);
    bf16_t* vtb  = (bf16_t*)take(32ull * 64 * 256 * 2);
    bf16_t* aout = (bf16_t*)take(16384ull * 512 * 2);

    cvt_kernel<<<32768, 256, 0, stream>>>(x,  xb,  16384 * 512);
    cvt_kernel<<<1024,  256, 0, stream>>>(Wq, wqb, 512 * 512);
    cvt_kernel<<<1024,  256, 0, stream>>>(Wk, wkb, 512 * 512);
    cvt_kernel<<<1024,  256, 0, stream>>>(Wv, wvb, 512 * 512);
    cvt_kernel<<<1024,  256, 0, stream>>>(Wp, wpb, 512 * 512);
    wsrT_kernel<<<16384, 256, 0, stream>>>(Wsr, wsrT);

    qproj_kernel <<<dim3(64, 8),  256, 0, stream>>>(xb, wqb, bq, qbuf);
    srconv_kernel<<<dim3(4, 8),   256, 0, stream>>>(xb, wsrT, bsr, cbuf);
    ln_kernel    <<<1024,         256, 0, stream>>>(cbuf, gamma, beta, xkv);
    kvproj_kernel<<<dim3(4, 8),   256, 0, stream>>>(xkv, wkb, bk, kbuf, nullptr, 0);
    kvproj_kernel<<<dim3(4, 8),   256, 0, stream>>>(xkv, wvb, bv, nullptr, vtb, 1);
    attn_kernel  <<<dim3(32, 32), 256, 0, stream>>>(qbuf, kbuf, vtb, aout);
    oproj_kernel <<<dim3(64, 8),  256, 0, stream>>>(aout, wpb, bp, out);
}
